// MySimpleRNN_85091892068982
// MI455X (gfx1250) — matrix operand order repaired and checked by reading, not compiled, not run on hardware
//
#include <hip/hip_runtime.h>
#include <hip/hip_bf16.h>
#include <math.h>

// ---------------------------------------------------------------------------
// MySimpleRNN on gfx1250 (MI455X):
//   Phase 0: transpose + cvt W_xh, W_hh -> bf16, by output column (in ws)
//   Phase 1: xW = inputs @ W_xh + b  via v_wmma_f32_16x16x32_bf16.
//            Wxt staged into LDS via global_load_async_to_lds_b128, B frags
//            double-buffered from LDS, A fragments pipelined from global.
//   Phase 2: sequential scan h_t = tanh(xW_t + h_{t-1} @ W_hh). W_hh lives in
//            VGPR fragments for all 1024 steps, h ping-pongs between two LDS
//            buffers (one barrier per step). d_out overwritten in place.
// ---------------------------------------------------------------------------

typedef __attribute__((ext_vector_type(16))) __bf16 bf16x16;
typedef __attribute__((ext_vector_type(8)))  __bf16 bf16x8;
typedef __attribute__((ext_vector_type(8)))  float  f32x8;
typedef __attribute__((ext_vector_type(4)))  int    i32x4;

constexpr int kT = 1024;
constexpr int kD = 256;
constexpr int kU = 256;
constexpr int kHS = 272;   // LDS row stride for h (bf16 elems): 544B, 16B-aligned

// ---------------- Phase 0: weight transpose + bf16 convert ------------------
__global__ void __launch_bounds__(256)
rnn_prep_weights(const float* __restrict__ Wxh, const float* __restrict__ Whh,
                 __bf16* __restrict__ Wxt, __bf16* __restrict__ Wht) {
  int idx = blockIdx.x * 256 + threadIdx.x;      // 0 .. 2*65536-1
  int which = idx >> 16;
  int e = idx & 0xFFFF;
  int n = e >> 8;
  int k = e & 255;
  if (which == 0) Wxt[e] = (__bf16)Wxh[k * 256 + n];
  else            Wht[e] = (__bf16)Whh[k * 256 + n];
}

// ---------------- Phase 1: input projection GEMM ----------------------------
// 512 blocks x 256 threads (8 waves). Each wave computes a 16x256 strip.
// Whole Wxt (128 KB) is staged into LDS once per block.
__global__ void __launch_bounds__(256)
rnn_xproj(const float* __restrict__ x,      // [B*T, 256] f32
          const __bf16* __restrict__ Wxt,   // [256 (n)][256 (k)] bf16
          const float* __restrict__ bias,   // [256]
          float* __restrict__ out)          // [B*T, 256] f32
{
  __shared__ __attribute__((aligned(16))) __bf16 wlds[256 * 256];  // 128 KB

  const int tid  = threadIdx.x;
  const int lane = tid & 31;
  const int wave = tid >> 5;
  const int lm = lane & 15;                 // row (A) / col (B,C) within tile
  const int lh = lane >> 4;                 // half-wave select
  const int row0 = (blockIdx.x * 8 + wave) * 16;

  // ---- Stage Wxt -> LDS (8192 16B chunks, 32 per thread) ----
#if __has_builtin(__builtin_amdgcn_global_load_async_to_lds_b128)
#pragma unroll
  for (int i = 0; i < 32; ++i) {
    const size_t boff = ((size_t)(i * 256 + tid)) * 16;
    __builtin_amdgcn_global_load_async_to_lds_b128(
        (__attribute__((address_space(1))) i32x4*)((char*)Wxt + boff),
        (__attribute__((address_space(3))) i32x4*)((char*)wlds + boff),
        0, 0);
  }
#if __has_builtin(__builtin_amdgcn_s_wait_asynccnt)
  __builtin_amdgcn_s_wait_asynccnt(0);
#else
  asm volatile("s_wait_asynccnt 0x0" ::: "memory");
#endif
#else
  for (int i = 0; i < 32; ++i) {
    const size_t c = (size_t)(i * 256 + tid) * 16;
    *(uint4*)((char*)wlds + c) = *(const uint4*)((const char*)Wxt + c);
  }
#endif
  __syncthreads();

  // ---- Main GEMM: acc[16] tiles, A pipelined, B double-buffered from LDS ---
  f32x8 acc[16] = {};
  const float* arow = x + (size_t)(row0 + lm) * kD + lh * 16;
  f32x8 alo = *(const f32x8*)(arow);
  f32x8 ahi = *(const f32x8*)(arow + 8);

#pragma unroll
  for (int k = 0; k < 8; ++k) {
    const int k0 = k * 32;
    bf16x16 a;
#pragma unroll
    for (int i = 0; i < 8; ++i) { a[i] = (__bf16)alo[i]; a[8 + i] = (__bf16)ahi[i]; }
    if (k < 7) {  // fetch next k's A while this k computes
      alo = *(const f32x8*)(arow + (k + 1) * 32);
      ahi = *(const f32x8*)(arow + (k + 1) * 32 + 8);
    }
    // B fragment: 16 contiguous bf16 of column n*16+lm, from LDS
    const __bf16* bbase = wlds + (size_t)lm * kD + k0 + lh * 16;
    bf16x16 bcur = *(const bf16x16*)(bbase);
    bf16x16 bnxt = *(const bf16x16*)(bbase + 16 * kD);
#pragma unroll
    for (int n = 0; n < 16; ++n) {
      bf16x16 bpre = bcur;
      if (n < 14) bpre = *(const bf16x16*)(bbase + (size_t)(n + 2) * 16 * kD);
      acc[n] = __builtin_amdgcn_wmma_f32_16x16x32_bf16(
          false, a, false, bcur, (short)0, acc[n], false, false);
      bcur = bnxt;
      bnxt = bpre;
    }
  }

  // Epilogue: + bias, store f32. C layout: M = r + 8*lh, N = lm.
#pragma unroll
  for (int n = 0; n < 16; ++n) {
    const float bv = bias[n * 16 + lm];
    float* op = out + (size_t)(row0 + lh * 8) * kU + n * 16 + lm;
#pragma unroll
    for (int r = 0; r < 8; ++r) op[(size_t)r * kU] = acc[n][r] + bv;
  }
}

// ---------------- Phase 2: sequential recurrence ----------------------------
// 4 blocks x 256 threads. Block owns batch rows [b0, b0+16). Wave w owns
// output columns [32w, 32w+32). h ping-pongs between two LDS buffers so only
// ONE barrier per timestep is needed.
__global__ void __launch_bounds__(256)
rnn_scan(const __bf16* __restrict__ Wht,   // [256 (n)][256 (k)] bf16
         float* __restrict__ hx)           // [B, T, U]: in = xW, out = h
{
  __shared__ __attribute__((aligned(16))) __bf16 hb[2 * 16 * kHS];

  const int lane = threadIdx.x & 31;
  const int wave = threadIdx.x >> 5;
  const int lm = lane & 15;
  const int lh = lane >> 4;
  const int b0 = blockIdx.x * 16;
  const int n0 = wave * 32;                // first owned output column

  // W_hh B-fragments, resident for the whole scan: 2 x 8 x 8 = 128 VGPRs.
  bf16x16 bw[2][8];
#pragma unroll
  for (int nt = 0; nt < 2; ++nt)
#pragma unroll
    for (int k = 0; k < 8; ++k)
      bw[nt][k] = *(const bf16x16*)(Wht + (size_t)(n0 + nt * 16 + lm) * kU + k * 32 + lh * 16);

  __bf16* hcur = hb;                 // h_{t-1}
  __bf16* hnxt = hb + 16 * kHS;      // h_t
  for (int i = threadIdx.x; i < 16 * kHS; i += 256) hcur[i] = (__bf16)0.0f;
  __syncthreads();

  for (int t = 0; t < kT; ++t) {
    // xW for this step (same addresses are overwritten with h_t below).
    float xw[2][8];
#pragma unroll
    for (int nt = 0; nt < 2; ++nt) {
      const int col = n0 + nt * 16 + lm;
#pragma unroll
      for (int r = 0; r < 8; ++r)
        xw[nt][r] = hx[((size_t)(b0 + lh * 8 + r) * kT + t) * kU + col];
    }
    if (t + 1 < kT)  // warm next step's cachelines (global_prefetch_b8)
      __builtin_prefetch(&hx[((size_t)(b0 + lh * 8) * kT + (t + 1)) * kU + n0 + lm], 0, 0);

    // All 8 A-fragments of h_{t-1} up front, then dense WMMA chains.
    bf16x16 afr[8];
#pragma unroll
    for (int k = 0; k < 8; ++k) {
      const __bf16* ap = hcur + lm * kHS + k * 32 + lh * 16;
      bf16x8 lo = *(const bf16x8*)ap;
      bf16x8 hi = *(const bf16x8*)(ap + 8);
#pragma unroll
      for (int i = 0; i < 8; ++i) { afr[k][i] = lo[i]; afr[k][8 + i] = hi[i]; }
    }
    f32x8 c0 = {}, c1 = {};
#pragma unroll
    for (int k = 0; k < 8; ++k) {
      c0 = __builtin_amdgcn_wmma_f32_16x16x32_bf16(false, afr[k], false, bw[0][k], (short)0, c0, false, false);
      c1 = __builtin_amdgcn_wmma_f32_16x16x32_bf16(false, afr[k], false, bw[1][k], (short)0, c1, false, false);
    }

    float h0[8], h1[8];
#pragma unroll
    for (int r = 0; r < 8; ++r) {
      h0[r] = tanhf(c0[r] + xw[0][r]);
      h1[r] = tanhf(c1[r] + xw[1][r]);
    }

    // Write h_t into the OTHER buffer (no WAR hazard) + in-place to global.
#pragma unroll
    for (int r = 0; r < 8; ++r) {
      const int row = lh * 8 + r;
      hnxt[row * kHS + n0 + lm]      = (__bf16)h0[r];
      hnxt[row * kHS + n0 + 16 + lm] = (__bf16)h1[r];
      hx[((size_t)(b0 + row) * kT + t) * kU + n0 + lm]      = h0[r];
      hx[((size_t)(b0 + row) * kT + t) * kU + n0 + 16 + lm] = h1[r];
    }
    __syncthreads();   // h_t visible to all waves; next step reads hnxt
    __bf16* tmp = hcur; hcur = hnxt; hnxt = tmp;
  }
}

// ---------------------------------------------------------------------------
extern "C" void kernel_launch(void* const* d_in, const int* in_sizes, int n_in,
                              void* d_out, int out_size, void* d_ws, size_t ws_size,
                              hipStream_t stream) {
  const float* x   = (const float*)d_in[0];   // [64,1024,256]
  const float* Wxh = (const float*)d_in[1];   // [256,256]
  const float* Whh = (const float*)d_in[2];   // [256,256]
  const float* bh  = (const float*)d_in[3];   // [256]
  float* out = (float*)d_out;                 // [64,1024,256]

  __bf16* Wxt = (__bf16*)d_ws;                // 128 KB
  __bf16* Wht = Wxt + 256 * 256;              // 128 KB

  rnn_prep_weights<<<512, 256, 0, stream>>>(Wxh, Whh, Wxt, Wht);
  rnn_xproj<<<512, 256, 0, stream>>>(x, Wxt, bh, out);
  rnn_scan<<<4, 256, 0, stream>>>(Wht, out);
}
